// MultiHeadAttention_18287970746820
// MI455X (gfx1250) — matrix operand order repaired and checked by reading, not compiled, not run on hardware
//
#include <hip/hip_runtime.h>
#include <hip/hip_bf16.h>

// ---- problem constants (match reference) ----
#define VOCAB   32000
#define DMODEL  1024
#define NHEADS  16
#define DK      64
#define SEQ     2048
#define BATCH   2
#define TOKENS  (BATCH * SEQ)   // 4096

typedef __attribute__((ext_vector_type(16))) __bf16 v16bf;
typedef __attribute__((ext_vector_type(8)))  __bf16 v8bf;
typedef __attribute__((ext_vector_type(8)))  float  v8f;

union Frag { v16bf v; v8bf h[2]; };

// ---------------- helpers ----------------
__device__ __forceinline__ __bf16 f2bf(float f) {
  unsigned u = __builtin_bit_cast(unsigned, f);
  u += 0x7FFFu + ((u >> 16) & 1u);          // round-to-nearest-even
  unsigned short s = (unsigned short)(u >> 16);
  return __builtin_bit_cast(__bf16, s);
}

// Fragment stored packed fragment-major: [lane][16 elems] (512 bf16 / tile).
__device__ __forceinline__ v16bf load_frag_packed(const __bf16* p) {
  const __bf16* q = p + (threadIdx.x & 31) * 16;
  Frag f;
  f.h[0] = *(const v8bf*)(q);
  f.h[1] = *(const v8bf*)(q + 8);
  return f.v;
}

// A fragment 16x32 (MxK) from row-major memory: lane L holds M=L%16,
// K in {kh..kh+7} u {16+kh..16+kh+7}, kh=8*(L/16) -> two 128-bit runs.
__device__ __forceinline__ v16bf load_a_rowmajor(const __bf16* base, int ld) {
  const int lane = threadIdx.x & 31;
  const int M    = lane & 15;
  const int kh   = (lane >> 4) << 3;
  const __bf16* p = base + (size_t)M * ld + kh;
  Frag f;
  f.h[0] = *(const v8bf*)(p);
  f.h[1] = *(const v8bf*)(p + 16);
  return f.v;
}

// slot of element (M,K) inside a packed A fragment (K in [0,32))
__device__ __forceinline__ int a_slot(int M, int K) {
  const int lane = M + (((K >> 3) & 1) << 4);
  const int i    = ((K >> 4) << 3) + (K & 7);
  return lane * 16 + i;
}
// slot of element (K,N) inside a packed B fragment (K in [0,32))
__device__ __forceinline__ int b_slot(int K, int N) {
  const int lane = N + (((K >> 3) & 1) << 4);
  const int i    = ((K >> 4) << 3) + (K & 7);
  return lane * 16 + i;
}

__device__ __forceinline__ v8f wmma_bf16(v16bf a, v16bf b, v8f c) {
  return __builtin_amdgcn_wmma_f32_16x16x32_bf16(
      false, a, false, b, (short)0, c, false, false);
}

// ---------------- kernels ----------------

__global__ __launch_bounds__(256) void k_embed(
    const int* __restrict__ x, const float* __restrict__ emb,
    __bf16* __restrict__ h) {
  const int tok = blockIdx.x;
  const int row = x[tok];
  const float4 v = ((const float4*)(emb + (size_t)row * DMODEL))[threadIdx.x];
  __bf16* dst = h + (size_t)tok * DMODEL + threadIdx.x * 4;
  dst[0] = f2bf(v.x); dst[1] = f2bf(v.y); dst[2] = f2bf(v.z); dst[3] = f2bf(v.w);
}

// Pack weight (1024x1024 f32, KxN) into B-fragment-major bf16.
__global__ __launch_bounds__(256) void k_packw(
    const float* __restrict__ w, __bf16* __restrict__ wp) {
  const int e = blockIdx.x * blockDim.x + threadIdx.x;   // 1M threads
  const int k = e >> 10, n = e & 1023;
  const int tile = ((k >> 5) << 6) + (n >> 4);
  wp[(size_t)tile * 512 + b_slot(k & 31, n & 15)] = f2bf(w[e]);
}

// GEMM: P = h(4096x1024) * w + bias. One wave -> 16(M) x 128(N) tile.
// grid = 256 blocks x 256 threads.
// MODE 0: Q*0.125 as packed B-frags [bh][stile][kstep]   (B operand of K*Q^T)
// MODE 1: K       as packed A-frags [bh][ttile][kstep]   (A operand of K*Q^T)
// MODE 2: V       as packed B-frags [bh][t32][ntile]     (B operand of P*V)
template <int MODE>
__global__ __launch_bounds__(256) void k_proj(
    const __bf16* __restrict__ h, const __bf16* __restrict__ wp,
    const float* __restrict__ bias, __bf16* __restrict__ dst) {
  const int wave = threadIdx.x >> 5;
  const int lane = threadIdx.x & 31;
  const int job  = blockIdx.x * 8 + wave;           // 2048 wave-jobs
  const int m0   = (job >> 3) << 4;
  const int n0   = (job & 7) << 7;
  v8f acc[8] = {};
  for (int kk = 0; kk < DMODEL; kk += 32) {
    __builtin_prefetch(h + (size_t)m0 * DMODEL + kk + 64, 0, 1);
    const v16bf a = load_a_rowmajor(h + (size_t)m0 * DMODEL + kk, DMODEL);
    const __bf16* wt = wp + (size_t)(((kk >> 5) << 6) + (n0 >> 4)) * 512;
#pragma unroll
    for (int t = 0; t < 8; ++t)
      acc[t] = wmma_bf16(a, load_frag_packed(wt + t * 512), acc[t]);
  }
  const int rbase = (lane >> 4) << 3;
  const int ncol  = lane & 15;
#pragma unroll
  for (int t = 0; t < 8; ++t) {
#pragma unroll
    for (int r = 0; r < 8; ++r) {
      const int m  = m0 + rbase + r;                // token index
      const int n  = n0 + t * 16 + ncol;            // model column
      const float val = acc[t][r] + bias[n];
      const int b  = m >> 11, s = m & (SEQ - 1);
      const int hd = n >> 6, dk = n & 63;
      const int bh = b * NHEADS + hd;
      const size_t tile = ((size_t)bh * 128 + (s >> 4)) * 2 + (dk >> 5);
      if (MODE == 0) {        // Q as B fragments (K dim = dk, N dim = query)
        dst[tile * 512 + b_slot(dk & 31, s & 15)] = f2bf(val * 0.125f);
      } else if (MODE == 1) { // K as A fragments (M = key, K dim = dk)
        dst[tile * 512 + a_slot(s & 15, dk & 31)] = f2bf(val);
      } else {                // V as B fragments (K dim = key pos, N dim = dk)
        const size_t vt = ((size_t)bh * 64 + (s >> 5)) * 4 + (dk >> 4);
        dst[vt * 512 + b_slot(s & 31, dk & 15)] = f2bf(val);
      }
    }
  }
}

// Flash attention with TRANSPOSED scores: S^T = K * Q^T.
// C-layout of S^T: lane owns one query (N=lane%16); its 16 values at reg r are
// keys kh+r and 16+kh+r (kh=8*(lane/16)) == exactly the P A-fragment layout.
// grid = (SEQ/128, BATCH*NHEADS), 256 threads (8 waves, 16 queries each).
__global__ __launch_bounds__(256) void k_attn(
    const __bf16* __restrict__ qp, const __bf16* __restrict__ kp,
    const __bf16* __restrict__ vp, __bf16* __restrict__ o) {
  const int wave = threadIdx.x >> 5;
  const int lane = threadIdx.x & 31;
  const int bh   = blockIdx.y;
  const int s0   = blockIdx.x * 128 + wave * 16;
  const int b    = bh >> 4, head = bh & 15;

  const __bf16* qb = qp + (((size_t)bh * 128 + (s0 >> 4)) * 2) * 512;
  const v16bf bq0 = load_frag_packed(qb);           // dk 0..31
  const v16bf bq1 = load_frag_packed(qb + 512);     // dk 32..63

  float mx = -1e30f, lsum = 0.0f;                   // per-lane (query = lane%16)
  v8f oacc[4] = {};
  const int rbase = (lane >> 4) << 3;
  const int ncol  = lane & 15;

  for (int t0 = 0; t0 < SEQ; t0 += 32) {
    const __bf16* ka = kp + (((size_t)bh * 128 + (t0 >> 4)) * 2) * 512;
    __builtin_prefetch(ka + 2048, 0, 1);
    v8f sc0 = {}, sc1 = {};                         // keys t0..+15 / t0+16..+31
    sc0 = wmma_bf16(load_frag_packed(ka), bq0, sc0);
    sc0 = wmma_bf16(load_frag_packed(ka + 512), bq1, sc0);
    sc1 = wmma_bf16(load_frag_packed(ka + 1024), bq0, sc1);
    sc1 = wmma_bf16(load_frag_packed(ka + 1536), bq1, sc1);

    // per-query max over 32 keys: 15 in-lane max + one cross-half exchange
    float tmax = fmaxf(sc0[0], sc1[0]);
#pragma unroll
    for (int i = 1; i < 8; ++i) tmax = fmaxf(tmax, fmaxf(sc0[i], sc1[i]));
    tmax = fmaxf(tmax, __shfl_xor(tmax, 16, 32));
    const float nm    = fmaxf(mx, tmax);
    const float scale = __expf(mx - nm);
    mx = nm;

    Frag pf;                                        // P tile, already A-layout
    float rs = 0.0f;
#pragma unroll
    for (int i = 0; i < 8; ++i) {
      const float p0 = __expf(sc0[i] - nm);
      const float p1 = __expf(sc1[i] - nm);
      rs += p0 + p1;
      pf.v[i]     = f2bf(p0);
      pf.v[i + 8] = f2bf(p1);
    }
    rs += __shfl_xor(rs, 16, 32);
    lsum = lsum * scale + rs;

    // rescale O rows (row query = rbase+r; scale lives at lane query)
#pragma unroll
    for (int r = 0; r < 8; ++r) {
      const float srow = __shfl(scale, rbase + r, 16);
#pragma unroll
      for (int t = 0; t < 4; ++t) oacc[t][r] *= srow;
    }

    const __bf16* vb = vp + (((size_t)bh * 64 + (t0 >> 5)) * 4) * 512;
#pragma unroll
    for (int t = 0; t < 4; ++t)
      oacc[t] = wmma_bf16(pf.v, load_frag_packed(vb + t * 512), oacc[t]);
  }

  // normalize (per-row 1/l) and scatter to o (B,S,DMODEL) row-major
#pragma unroll
  for (int r = 0; r < 8; ++r) {
    const float linv = 1.0f / __shfl(lsum, rbase + r, 16);
    const int srow = s0 + rbase + r;
#pragma unroll
    for (int t = 0; t < 4; ++t) {
      o[((size_t)b * SEQ + srow) * DMODEL + head * DK + t * 16 + ncol] =
          f2bf(oacc[t][r] * linv);
    }
  }
}

// out = o(4096x1024) * wo + bo, f32 output. One wave -> 16x128 tile.
__global__ __launch_bounds__(256) void k_out(
    const __bf16* __restrict__ oi, const __bf16* __restrict__ wp,
    const float* __restrict__ bias, float* __restrict__ dst) {
  const int wave = threadIdx.x >> 5;
  const int lane = threadIdx.x & 31;
  const int job  = blockIdx.x * 8 + wave;
  const int m0   = (job >> 3) << 4;
  const int n0   = (job & 7) << 7;
  v8f acc[8] = {};
  for (int kk = 0; kk < DMODEL; kk += 32) {
    __builtin_prefetch(oi + (size_t)m0 * DMODEL + kk + 64, 0, 1);
    const v16bf a = load_a_rowmajor(oi + (size_t)m0 * DMODEL + kk, DMODEL);
    const __bf16* wt = wp + (size_t)(((kk >> 5) << 6) + (n0 >> 4)) * 512;
#pragma unroll
    for (int t = 0; t < 8; ++t)
      acc[t] = wmma_bf16(a, load_frag_packed(wt + t * 512), acc[t]);
  }
  const int rbase = (lane >> 4) << 3;
  const int ncol  = lane & 15;
#pragma unroll
  for (int t = 0; t < 8; ++t) {
#pragma unroll
    for (int r = 0; r < 8; ++r) {
      const int m = m0 + rbase + r;
      const int n = n0 + t * 16 + ncol;
      dst[(size_t)m * DMODEL + n] = acc[t][r] + bias[n];
    }
  }
}

// ---------------- launch ----------------
extern "C" void kernel_launch(void* const* d_in, const int* in_sizes, int n_in,
                              void* d_out, int out_size, void* d_ws, size_t ws_size,
                              hipStream_t stream) {
  (void)in_sizes; (void)n_in; (void)out_size; (void)ws_size;
  const int*   x   = (const int*)d_in[0];
  const float* emb = (const float*)d_in[1];
  const float* wq  = (const float*)d_in[2];
  const float* bq  = (const float*)d_in[3];
  const float* wk  = (const float*)d_in[4];
  const float* bk  = (const float*)d_in[5];
  const float* wv  = (const float*)d_in[6];
  const float* bv  = (const float*)d_in[7];
  const float* wo  = (const float*)d_in[8];
  const float* bo  = (const float*)d_in[9];
  float* out = (float*)d_out;

  char* ws = (char*)d_ws;
  const size_t MB = (size_t)1 << 20;
  __bf16* hb  = (__bf16*)(ws +  0 * MB);   // 4096x1024 row-major bf16 (8 MiB)
  __bf16* wqp = (__bf16*)(ws +  8 * MB);   // packed weight B-frags (2 MiB each)
  __bf16* wkp = (__bf16*)(ws + 10 * MB);
  __bf16* wvp = (__bf16*)(ws + 12 * MB);
  __bf16* wop = (__bf16*)(ws + 14 * MB);
  __bf16* qp  = (__bf16*)(ws + 16 * MB);   // Q packed B-frags (8 MiB)
  __bf16* kp  = (__bf16*)(ws + 24 * MB);   // K packed A-frags (8 MiB)
  __bf16* vp  = (__bf16*)(ws + 32 * MB);   // V packed B-frags (8 MiB)
  __bf16* ob  = (__bf16*)(ws + 40 * MB);   // (B,S,D) row-major (8 MiB)

  k_embed<<<TOKENS, 256, 0, stream>>>(x, emb, hb);
  k_packw<<<4096, 256, 0, stream>>>(wq, wqp);
  k_packw<<<4096, 256, 0, stream>>>(wk, wkp);
  k_packw<<<4096, 256, 0, stream>>>(wv, wvp);
  k_packw<<<4096, 256, 0, stream>>>(wo, wop);

  k_proj<0><<<256, 256, 0, stream>>>(hb, wqp, bq, qp);
  k_proj<1><<<256, 256, 0, stream>>>(hb, wkp, bk, kp);
  k_proj<2><<<256, 256, 0, stream>>>(hb, wvp, bv, vp);

  k_attn<<<dim3(SEQ / 128, BATCH * NHEADS), 256, 0, stream>>>(qp, kp, vp, ob);

  k_out<<<256, 256, 0, stream>>>(ob, wop, bo, out);
}
